// VariationalEncoder_15315853378095
// MI455X (gfx1250) — matrix operand order repaired and checked by reading, not compiled, not run on hardware
//
#include <hip/hip_runtime.h>
#include <math.h>

typedef _Float16 half_t;
typedef __attribute__((ext_vector_type(16))) _Float16 v16h;
typedef __attribute__((ext_vector_type(8)))  float    v8f;
typedef __attribute__((ext_vector_type(4)))  unsigned int v4u;
typedef __attribute__((ext_vector_type(8)))  int v8i;
typedef __attribute__((ext_vector_type(4)))  int v4i;

#if defined(__has_builtin)
#if __has_builtin(__builtin_amdgcn_tensor_load_to_lds) && \
    __has_builtin(__builtin_amdgcn_s_wait_tensorcnt)
#define USE_TDM 1
#endif
#endif

#define BSZ  64
#define TT   128
#define WIN  318
#define C1N  24
#define F1   288    // conv1 out width
#define F1P  286    // pooled width
#define TP   130    // t rows incl pads
#define F2   256    // conv2 out width
#define K2   2232   // 24*3*31
#define K2P  2240   // padded to mult of 32

__device__ __forceinline__ float siluf(float x) { return x / (1.0f + __expf(-x)); }
__device__ __forceinline__ float sigf(float x)  { return 1.0f / (1.0f + __expf(-x)); }
__device__ __forceinline__ v8f zero8() {
    v8f z;
#pragma unroll
    for (int i = 0; i < 8; ++i) z[i] = 0.0f;
    return z;
}

#ifdef USE_TDM
// TDM: async-load a [rows x 32] f16 tile (row stride = ldK halfs) from gptr
// into LDS at byte offset lds_off.  D# per CDNA5 ISA ch.8 (2D, groups 2/3 null).
__device__ __forceinline__ void tdm_load_tile(const half_t* gptr, unsigned lds_off,
                                              int ldK, int Mtot, int rows) {
    unsigned long long ga = (unsigned long long)(size_t)gptr;
    v4u g0;
    g0[0] = 1u;                                         // count=1, user descriptor
    g0[1] = lds_off;                                    // lds_addr (bytes)
    g0[2] = (unsigned)(ga & 0xffffffffull);             // global_addr[31:0]
    g0[3] = (unsigned)((ga >> 32) & 0x01ffffffull)      // global_addr[56:32]
            | (2u << 30);                               // type = 2 ("image")
    unsigned d0 = (unsigned)ldK;                        // tensor_dim0 (elements)
    unsigned d1 = (unsigned)Mtot;                       // tensor_dim1
    v8i g1;
    g1[0] = (int)(1u << 16);                            // data_size=1 (2B), mask=0
    g1[1] = (int)((d0 & 0xffffu) << 16);                // tensor_dim0 lo16
    g1[2] = (int)((d0 >> 16) | ((d1 & 0xffffu) << 16)); // dim0 hi / dim1 lo
    g1[3] = (int)((d1 >> 16) | (32u << 16));            // dim1 hi / tile_dim0=32
    g1[4] = rows;                                       // tile_dim1, tile_dim2=0
    g1[5] = (int)d0;                                    // tensor_dim0_stride lo32
    g1[6] = 0;                                          // stride hi / dim1_stride lo
    g1[7] = 0;
    v4i z4 = {0, 0, 0, 0};
#if __clang_major__ >= 23
    v8i z8 = {0, 0, 0, 0, 0, 0, 0, 0};
    __builtin_amdgcn_tensor_load_to_lds(g0, g1, z4, z4, z8, 0);
#else
    __builtin_amdgcn_tensor_load_to_lds(g0, g1, z4, z4, 0);
#endif
}
#endif

// ---------------------------------------------------------------------------
// init: maxpool pads (-inf surrogate), conv2 zero t-pads, stat accumulators
// ---------------------------------------------------------------------------
__global__ void k_init(half_t* P, half_t* Y, float* stats) {
    const int nP = BSZ * C1N * 2 * F1;   // 884736
    const int nY = BSZ * C1N * 2 * F1P;  // 878592
    int idx = blockIdx.x * 256 + threadIdx.x;
    if (idx < nP) {
        int f = idx % F1; int i = idx / F1;
        int side = i & 1; i >>= 1;
        int c = i % C1N; int b = i / C1N;
        int t = side ? (TP - 1) : 0;
        P[(((b * C1N + c) * TP) + t) * F1 + f] = (half_t)(-60000.0f);
    } else if (idx < nP + nY) {
        int j = idx - nP;
        int f = j % F1P; int i = j / F1P;
        int side = i & 1; i >>= 1;
        int c = i % C1N; int b = i / C1N;
        int t = side ? (TP - 1) : 0;
        Y[(((b * C1N + c) * TP) + t) * F1P + f] = (half_t)0.0f;
    } else if (idx < nP + nY + 64) {
        stats[idx - nP - nY] = 0.0f;
    }
}

// ---------------------------------------------------------------------------
// weight packing: conv2 -> f16 [K2P][48] (im2col-K-major); generic W^T pad->f16
// ---------------------------------------------------------------------------
__global__ void k_pack_w2(const float* w2, half_t* Wp2) {
    int idx = blockIdx.x * 256 + threadIdx.x;
    if (idx >= K2P * 48) return;
    int n = idx % 48, k = idx / 48;
    float v = 0.0f;
    if (k < K2) {
        int c = k / 93, r = k - c * 93;
        int dt = r / 31, df = r - dt * 31;
        v = w2[((n * C1N + c) * 3 + dt) * 31 + df];
    }
    Wp2[k * 48 + n] = (half_t)v;
}

__global__ void k_pack_T(const float* W, half_t* out, int N, int K, int Ntot) {
    int idx = blockIdx.x * 256 + threadIdx.x;
    if (idx >= K * Ntot) return;
    int n = idx % Ntot, k = idx / Ntot;
    out[idx] = (half_t)((n < N) ? W[n * K + k] : 0.0f);
}

// ---------------------------------------------------------------------------
// conv1 + SiLU  -> P f16 [B][24][130][288]  (rows 1..128 real)
// ---------------------------------------------------------------------------
__global__ __launch_bounds__(256) void k_conv1(const float* x, const float* w1,
                                               const float* b1, half_t* P) {
    __shared__ float ws_[C1N * 93];
    __shared__ float bs_[C1N];
    for (int i = threadIdx.x; i < C1N * 93; i += 256) ws_[i] = w1[i];
    if (threadIdx.x < C1N) bs_[threadIdx.x] = b1[threadIdx.x];
    __syncthreads();
    int idx = blockIdx.x * 256 + threadIdx.x;  // grid sized exactly
    int f = idx % F1; int i = idx / F1;
    int t = i % TT; i /= TT;
    int ch = i % C1N; int b = i / C1N;
    const float* xp = x + (long)b * TT * WIN;
    const float* wp = ws_ + ch * 93;
    float acc = bs_[ch];
    for (int dt = 0; dt < 3; ++dt) {
        int t2 = t + dt - 1;
        if ((unsigned)t2 < (unsigned)TT) {
            const float* xr = xp + t2 * WIN + f;
            const float* wr = wp + dt * 31;
#pragma unroll
            for (int df = 0; df < 31; ++df) acc += xr[df] * wr[df];
        }
    }
    P[(((b * C1N + ch) * TP) + (t + 1)) * F1 + f] = (half_t)siluf(acc);
}

// 3x3 maxpool at (b,ch,t,f) over P (t storage rows t..t+2, cols f..f+2)
__device__ __forceinline__ float pool9(const half_t* P, int b, int ch, int t, int f) {
    const half_t* base = P + (((b * C1N + ch) * TP) + t) * F1 + f;
    float m = -1e30f;
#pragma unroll
    for (int dr = 0; dr < 3; ++dr) {
        const half_t* r = base + dr * F1;
        m = fmaxf(m, fmaxf(fmaxf((float)r[0], (float)r[1]), (float)r[2]));
    }
    return m;
}

// ---------------------------------------------------------------------------
// BN1 batch statistics over pooled values:  grid(64,24)
// ---------------------------------------------------------------------------
__global__ __launch_bounds__(256) void k_stats1(const half_t* P, float* stats) {
    __shared__ float rs[256], rq[256];
    int b = blockIdx.x, ch = blockIdx.y, tid = threadIdx.x;
    float s = 0.0f, q = 0.0f;
    for (int i = tid; i < TT * F1P; i += 256) {
        int t = i / F1P, f = i % F1P;
        float v = pool9(P, b, ch, t, f);
        s += v; q += v * v;
    }
    rs[tid] = s; rq[tid] = q;
    __syncthreads();
    for (int m = 128; m > 0; m >>= 1) {
        if (tid < m) { rs[tid] += rs[tid + m]; rq[tid] += rq[tid + m]; }
        __syncthreads();
    }
    if (tid == 0) {
        atomicAdd(&stats[ch], rs[0]);
        atomicAdd(&stats[C1N + ch], rq[0]);
    }
}

// ---------------------------------------------------------------------------
// pool (recomputed, L2-hot) + BN1 -> Y f16 [B][24][130][286] (rows 1..128)
// ---------------------------------------------------------------------------
__global__ __launch_bounds__(256) void k_bn1apply(const half_t* P, const float* stats,
                                                  const float* g1, const float* bb1,
                                                  half_t* Y) {
    int idx = blockIdx.x * 256 + threadIdx.x;  // exact grid
    int f = idx % F1P; int i = idx / F1P;
    int t = i % TT; i /= TT;
    int ch = i % C1N; int b = i / C1N;
    float n = (float)(BSZ * TT * F1P);
    float mean = stats[ch] / n;
    float var = stats[C1N + ch] / n - mean * mean;
    float sc = g1[ch] * rsqrtf(var + 1e-5f);
    float sh = bb1[ch] - mean * sc;
    float v = pool9(P, b, ch, t, f) * sc + sh;
    Y[(((b * C1N + ch) * TP) + (t + 1)) * F1P + f] = (half_t)v;
}

// ---------------------------------------------------------------------------
// conv2 (im2col WMMA GEMM, M-tile 128 x N=48) + SiLU + conv3 channel reduce
//   -> featpre f32 [B][T][256]
// ---------------------------------------------------------------------------
__global__ __launch_bounds__(256) void k_conv2(const half_t* Y, const half_t* Wp2,
                                               const float* w3, const float* b3,
                                               float* featpre) {
    __shared__ half_t A_s[128][32];
    __shared__ half_t B_sT[48][32];
    __shared__ int koff[32];
    __shared__ float w3s[48];
    int tid = threadIdx.x;
    if (tid < 48) w3s[tid] = w3[tid];

    int blk = blockIdx.x;
    int f0 = (blk & 1) * 128;
    int bt = blk >> 1;
    int b = bt >> 7, t = bt & 127;
    const half_t* Ybt = Y + (long)b * C1N * TP * F1P + t * F1P;

    int lane = tid & 31, wv = tid >> 5;
    int lh = lane & 15, hi = lane >> 4;
    v8f acc[3];
    acc[0] = zero8(); acc[1] = zero8(); acc[2] = zero8();

    for (int kc = 0; kc < K2P; kc += 32) {
        __syncthreads();
        if (tid < 32) {
            int k = kc + tid, off = -1;
            if (k < K2) {
                int c = k / 93, r = k - c * 93;
                int dt = r / 31, df = r - dt * 31;
                off = c * (TP * F1P) + dt * F1P + df;
            }
            koff[tid] = off;
        }
        if (tid == 0 && kc + 32 < K2P) {
            // L2-resident weight slab: hint next K-slice (global_prefetch_b8)
            __builtin_prefetch(&Wp2[(kc + 32) * 48], 0, 1);
        }
        __syncthreads();
        {   // A tile: 128 rows (f0+row), 32 im2col K cols (gather via koff)
            int row = tid >> 1, kk0 = (tid & 1) * 16;
            int f = f0 + row;
#pragma unroll
            for (int j = 0; j < 16; ++j) {
                int kk = kk0 + j, off = koff[kk];
                A_s[row][kk] = (off >= 0) ? Ybt[off + f] : (half_t)0.0f;
            }
        }
        for (int i = tid; i < 48 * 32; i += 256) {  // B tile transposed [n][k]
            int n = i >> 5, kk = i & 31;
            B_sT[n][kk] = Wp2[(kc + kk) * 48 + n];
        }
        __syncthreads();

        v16h a;
        int arow = wv * 16 + lh;
#pragma unroll
        for (int v = 0; v < 8; ++v) {
            int k0 = (v & 4) * 4 + hi * 8 + (v & 3) * 2;  // A 16x32 layout
            a[2 * v] = A_s[arow][k0];
            a[2 * v + 1] = A_s[arow][k0 + 1];
        }
#pragma unroll
        for (int nt = 0; nt < 3; ++nt) {
            v16h bm;
            int col = nt * 16 + lh;
#pragma unroll
            for (int v = 0; v < 8; ++v) {
                int k0 = (v & 4) * 4 + hi * 8 + (v & 3) * 2;  // B 32x16 layout
                bm[2 * v] = B_sT[col][k0];
                bm[2 * v + 1] = B_sT[col][k0 + 1];
            }
            acc[nt] = __builtin_amdgcn_wmma_f32_16x16x32_f16(
                false, a, false, bm, (short)0, acc[nt], false, false);
        }
    }

    // epilogue: silu + conv3 weighted channel sum, reduce across 16-lane halves
    float part[8];
#pragma unroll
    for (int i = 0; i < 8; ++i) {
        part[i] = siluf(acc[0][i]) * w3s[lh] +
                  siluf(acc[1][i]) * w3s[16 + lh] +
                  siluf(acc[2][i]) * w3s[32 + lh];
    }
#pragma unroll
    for (int m = 1; m < 16; m <<= 1) {
#pragma unroll
        for (int i = 0; i < 8; ++i) part[i] += __shfl_xor(part[i], m, 32);
    }
    if (lh == 0) {
        float bb = b3[0];
#pragma unroll
        for (int i = 0; i < 8; ++i) {
            int mrow = wv * 16 + hi * 8 + i;   // C layout: rows 0-7 half0, 8-15 half1
            featpre[((b * TT) + t) * F2 + f0 + mrow] = part[i] + bb;
        }
    }
}

// ---------------------------------------------------------------------------
// BN2 stats (single channel, 2.1M elems)
// ---------------------------------------------------------------------------
__global__ __launch_bounds__(256) void k_bn2stats(const float* fp, float* stats) {
    __shared__ float rs[256], rq[256];
    int tid = threadIdx.x;
    float s = 0.0f, q = 0.0f;
    int base = blockIdx.x * 2048;
#pragma unroll
    for (int j = 0; j < 8; ++j) {
        int i = base + j * 256 + tid;
        float v = fp[i];
        s += v; q += v * v;
    }
    rs[tid] = s; rq[tid] = q;
    __syncthreads();
    for (int m = 128; m > 0; m >>= 1) {
        if (tid < m) { rs[tid] += rs[tid + m]; rq[tid] += rq[tid + m]; }
        __syncthreads();
    }
    if (tid == 0) { atomicAdd(&stats[48], rs[0]); atomicAdd(&stats[49], rq[0]); }
}

// BN2 + SiLU -> feat f16 [T][B][256]  (m = t*64+b row-major for LSTM GEMMs)
__global__ __launch_bounds__(256) void k_bn2apply(const float* fp, const float* stats,
                                                  const float* g2, const float* bb2,
                                                  half_t* feat) {
    int idx = blockIdx.x * 256 + threadIdx.x;  // exact grid 2097152
    int f = idx % F2; int i = idx / F2;
    int t = i % TT; int b = i / TT;
    float n = (float)(BSZ * TT * F2);
    float mean = stats[48] / n;
    float var = stats[49] / n - mean * mean;
    float sc = g2[0] * rsqrtf(var + 1e-5f);
    float sh = bb2[0] - mean * sc;
    float v = siluf(fp[idx] * sc + sh);
    feat[((t * BSZ) + b) * F2 + f] = (half_t)v;
}

// ---------------------------------------------------------------------------
// generic WMMA GEMM:  C[M][Ntot] = A[M][K] * B[K][Ntot]  (f16 in, f32 out)
// grid(M/128, Ntot/16), block 256 (8 waves, one 16x16 tile each).
// A tile (128x32, row stride K) is moved by the Tensor Data Mover when
// available (one tensor_load_to_lds per K-step, issued by wave 0, completion
// via s_wait_tensorcnt); falls back to cooperative VALU loads otherwise.
// ---------------------------------------------------------------------------
__global__ __launch_bounds__(256) void k_gemm(const half_t* A, const half_t* B,
                                              float* C, int M, int K, int Ntot) {
    __shared__ half_t A_s[128][32];
    __shared__ half_t B_sT[16][32];
    int tid = threadIdx.x;
    int m0 = blockIdx.x * 128, n0 = blockIdx.y * 16;
    int lane = tid & 31, wv = tid >> 5;
    int lh = lane & 15, hi = lane >> 4;
#ifdef USE_TDM
    unsigned ldsA = (unsigned)((size_t)(&A_s[0][0]) & 0x7ffffu);
#endif
    v8f acc = zero8();
    for (int kc = 0; kc < K; kc += 32) {
        __syncthreads();
#ifdef USE_TDM
        if (wv == 0) {
            tdm_load_tile(A + (long)m0 * K + kc, ldsA, K, M, 128);
        }
#else
        {
            int row = tid >> 1, kk0 = (tid & 1) * 16;
            const half_t* Ar = A + (long)(m0 + row) * K + kc + kk0;
#pragma unroll
            for (int j = 0; j < 16; ++j) A_s[row][kk0 + j] = Ar[j];
        }
#endif
        for (int i = tid; i < 512; i += 256) {
            int n = i >> 5, kk = i & 31;
            B_sT[n][kk] = B[(long)(kc + kk) * Ntot + n0 + n];
        }
#ifdef USE_TDM
        if (wv == 0) {
            __builtin_amdgcn_s_wait_tensorcnt(0);
        }
#endif
        __syncthreads();
        v16h a, bm;
        int arow = wv * 16 + lh;
#pragma unroll
        for (int vi = 0; vi < 8; ++vi) {
            int k0 = (vi & 4) * 4 + hi * 8 + (vi & 3) * 2;
            a[2 * vi] = A_s[arow][k0];
            a[2 * vi + 1] = A_s[arow][k0 + 1];
        }
#pragma unroll
        for (int vi = 0; vi < 8; ++vi) {
            int k0 = (vi & 4) * 4 + hi * 8 + (vi & 3) * 2;
            bm[2 * vi] = B_sT[lh][k0];
            bm[2 * vi + 1] = B_sT[lh][k0 + 1];
        }
        acc = __builtin_amdgcn_wmma_f32_16x16x32_f16(
            false, a, false, bm, (short)0, acc, false, false);
    }
#pragma unroll
    for (int i = 0; i < 8; ++i) {
        int row = m0 + wv * 16 + hi * 8 + i;
        C[(long)row * Ntot + n0 + lh] = acc[i];
    }
}

// ---------------------------------------------------------------------------
// wide LSTM recurrence (hidden 110), single persistent workgroup
// ---------------------------------------------------------------------------
__global__ __launch_bounds__(1024) void k_wide_rec(const float* Xg, const float* whh,
                                                   const float* bih, const float* bhh,
                                                   float* hout) {
    __shared__ float h_s[BSZ * 110];
    int tid = threadIdx.x;
    for (int i = tid; i < BSZ * 110; i += 1024) h_s[i] = 0.0f;
    float creg[7], hn[7];
#pragma unroll
    for (int p = 0; p < 7; ++p) creg[p] = 0.0f;
    __syncthreads();
    for (int t = 0; t < TT; ++t) {
#pragma unroll
        for (int p = 0; p < 7; ++p) {
            int pair = tid + p * 1024;
            hn[p] = 0.0f;
            if (pair < BSZ * 110) {
                int b = pair / 110, jh = pair - b * 110;
                const float* hr = &h_s[b * 110];
                const float* xr = &Xg[(long)(t * BSZ + b) * 448];
                float gi = xr[jh] + bih[jh] + bhh[jh];
                float gf = xr[110 + jh] + bih[110 + jh] + bhh[110 + jh];
                float gg = xr[220 + jh] + bih[220 + jh] + bhh[220 + jh];
                float go = xr[330 + jh] + bih[330 + jh] + bhh[330 + jh];
                const float* wi = whh + (long)jh * 110;
                const float* wf = whh + (long)(110 + jh) * 110;
                const float* wg = whh + (long)(220 + jh) * 110;
                const float* wo = whh + (long)(330 + jh) * 110;
                for (int k = 0; k < 110; ++k) {
                    float hv = hr[k];
                    gi += hv * wi[k]; gf += hv * wf[k];
                    gg += hv * wg[k]; go += hv * wo[k];
                }
                float c = sigf(gf) * creg[p] + sigf(gi) * tanhf(gg);
                creg[p] = c;
                hn[p] = sigf(go) * tanhf(c);
            }
        }
        __syncthreads();
#pragma unroll
        for (int p = 0; p < 7; ++p) {
            int pair = tid + p * 1024;
            if (pair < BSZ * 110) h_s[pair] = hn[p];
        }
        __syncthreads();
    }
    for (int i = tid; i < BSZ * 110; i += 1024) hout[i] = h_s[i];
}

// ---------------------------------------------------------------------------
// deep LSTM stack recurrence (90/66/48/24), single persistent workgroup
// ---------------------------------------------------------------------------
__global__ __launch_bounds__(1024) void k_deep_rec(
    const float* Xg1,
    const float* w1hh, const float* b1ih, const float* b1hh,
    const float* w2ih, const float* w2hh, const float* b2ih, const float* b2hh,
    const float* w3ih, const float* w3hh, const float* b3ih, const float* b3hh,
    const float* w4ih, const float* w4hh, const float* b4ih, const float* b4hh,
    float* h4out) {
    __shared__ float h1[BSZ * 90];
    __shared__ float h2[BSZ * 66];
    __shared__ float h3[BSZ * 48];
    __shared__ float h4[BSZ * 24];
    int tid = threadIdx.x;
    for (int i = tid; i < BSZ * 90; i += 1024) h1[i] = 0.0f;
    for (int i = tid; i < BSZ * 66; i += 1024) h2[i] = 0.0f;
    for (int i = tid; i < BSZ * 48; i += 1024) h3[i] = 0.0f;
    for (int i = tid; i < BSZ * 24; i += 1024) h4[i] = 0.0f;
    float c1r[6] = {0, 0, 0, 0, 0, 0};
    float c2r[5] = {0, 0, 0, 0, 0};
    float c3r[3] = {0, 0, 0};
    float c4r[2] = {0, 0};
    __syncthreads();
    for (int t = 0; t < TT; ++t) {
        // ---- layer 1 (in: precomputed Xg1, hidden 90) ----
        float hn1[6];
#pragma unroll
        for (int p = 0; p < 6; ++p) {
            int pair = tid + p * 1024;
            hn1[p] = 0.0f;
            if (pair < BSZ * 90) {
                int b = pair / 90, jh = pair - b * 90;
                const float* xr = &Xg1[(long)(t * BSZ + b) * 368];
                float g4[4];
#pragma unroll
                for (int gt = 0; gt < 4; ++gt) {
                    int j = gt * 90 + jh;
                    float g = xr[j] + b1ih[j] + b1hh[j];
                    const float* wr = w1hh + (long)j * 90;
                    const float* hr = &h1[b * 90];
                    for (int k = 0; k < 90; ++k) g += hr[k] * wr[k];
                    g4[gt] = g;
                }
                float c = sigf(g4[1]) * c1r[p] + sigf(g4[0]) * tanhf(g4[2]);
                c1r[p] = c;
                hn1[p] = sigf(g4[3]) * tanhf(c);
            }
        }
        __syncthreads();
#pragma unroll
        for (int p = 0; p < 6; ++p) {
            int pair = tid + p * 1024;
            if (pair < BSZ * 90) h1[pair] = hn1[p];
        }
        __syncthreads();
        // ---- layer 2 (in: h1 new (90), hidden 66) ----
        float hn2[5];
#pragma unroll
        for (int p = 0; p < 5; ++p) {
            int pair = tid + p * 1024;
            hn2[p] = 0.0f;
            if (pair < BSZ * 66) {
                int b = pair / 66, jh = pair - b * 66;
                float g4[4];
#pragma unroll
                for (int gt = 0; gt < 4; ++gt) {
                    int j = gt * 66 + jh;
                    float g = b2ih[j] + b2hh[j];
                    const float* wr = w2ih + (long)j * 90;
                    const float* xr = &h1[b * 90];
                    for (int k = 0; k < 90; ++k) g += xr[k] * wr[k];
                    const float* vr = w2hh + (long)j * 66;
                    const float* hr = &h2[b * 66];
                    for (int k = 0; k < 66; ++k) g += hr[k] * vr[k];
                    g4[gt] = g;
                }
                float c = sigf(g4[1]) * c2r[p] + sigf(g4[0]) * tanhf(g4[2]);
                c2r[p] = c;
                hn2[p] = sigf(g4[3]) * tanhf(c);
            }
        }
        __syncthreads();
#pragma unroll
        for (int p = 0; p < 5; ++p) {
            int pair = tid + p * 1024;
            if (pair < BSZ * 66) h2[pair] = hn2[p];
        }
        __syncthreads();
        // ---- layer 3 (in: h2 new (66), hidden 48) ----
        float hn3[3];
#pragma unroll
        for (int p = 0; p < 3; ++p) {
            int pair = tid + p * 1024;
            hn3[p] = 0.0f;
            if (pair < BSZ * 48) {
                int b = pair / 48, jh = pair - b * 48;
                float g4[4];
#pragma unroll
                for (int gt = 0; gt < 4; ++gt) {
                    int j = gt * 48 + jh;
                    float g = b3ih[j] + b3hh[j];
                    const float* wr = w3ih + (long)j * 66;
                    const float* xr = &h2[b * 66];
                    for (int k = 0; k < 66; ++k) g += xr[k] * wr[k];
                    const float* vr = w3hh + (long)j * 48;
                    const float* hr = &h3[b * 48];
                    for (int k = 0; k < 48; ++k) g += hr[k] * vr[k];
                    g4[gt] = g;
                }
                float c = sigf(g4[1]) * c3r[p] + sigf(g4[0]) * tanhf(g4[2]);
                c3r[p] = c;
                hn3[p] = sigf(g4[3]) * tanhf(c);
            }
        }
        __syncthreads();
#pragma unroll
        for (int p = 0; p < 3; ++p) {
            int pair = tid + p * 1024;
            if (pair < BSZ * 48) h3[pair] = hn3[p];
        }
        __syncthreads();
        // ---- layer 4 (in: h3 new (48), hidden 24) ----
        float hn4[2];
#pragma unroll
        for (int p = 0; p < 2; ++p) {
            int pair = tid + p * 1024;
            hn4[p] = 0.0f;
            if (pair < BSZ * 24) {
                int b = pair / 24, jh = pair - b * 24;
                float g4[4];
#pragma unroll
                for (int gt = 0; gt < 4; ++gt) {
                    int j = gt * 24 + jh;
                    float g = b4ih[j] + b4hh[j];
                    const float* wr = w4ih + (long)j * 48;
                    const float* xr = &h3[b * 48];
                    for (int k = 0; k < 48; ++k) g += xr[k] * wr[k];
                    const float* vr = w4hh + (long)j * 24;
                    const float* hr = &h4[b * 24];
                    for (int k = 0; k < 24; ++k) g += hr[k] * vr[k];
                    g4[gt] = g;
                }
                float c = sigf(g4[1]) * c4r[p] + sigf(g4[0]) * tanhf(g4[2]);
                c4r[p] = c;
                hn4[p] = sigf(g4[3]) * tanhf(c);
            }
        }
        __syncthreads();
#pragma unroll
        for (int p = 0; p < 2; ++p) {
            int pair = tid + p * 1024;
            if (pair < BSZ * 24) h4[pair] = hn4[p];
        }
        __syncthreads();
    }
    for (int i = tid; i < BSZ * 24; i += 1024) h4out[i] = h4[i];
}

// ---------------------------------------------------------------------------
// head: fc layers + reparameterization -> d_out = [mu | z]
// ---------------------------------------------------------------------------
__global__ void k_head(const float* h4f, const float* hwf,
                       const float* dfcw, const float* dfcb,
                       const float* wfcw, const float* wfcb,
                       const float* l1w, const float* l1b,
                       const float* f1w, const float* f1b,
                       const float* f2w, const float* f2b,
                       const float* eps, float* out) {
    int b = threadIdx.x;
    if (b >= BSZ) return;
    float hc[44];
    for (int i = 0; i < 22; ++i) {
        float s = dfcb[i];
        for (int k = 0; k < 24; ++k) s += h4f[b * 24 + k] * dfcw[i * 24 + k];
        hc[i] = s;
    }
    for (int i = 0; i < 22; ++i) {
        float s = wfcb[i];
        for (int k = 0; k < 110; ++k) s += hwf[b * 110 + k] * wfcw[i * 110 + k];
        hc[22 + i] = s;
    }
    float hl[40];
    for (int j = 0; j < 40; ++j) {
        float s = l1b[j];
        for (int k = 0; k < 44; ++k) s += hc[k] * l1w[j * 44 + k];
        hl[j] = s;
    }
    for (int o = 0; o < 32; ++o) {
        float mu = f1b[o], lg = f2b[o];
        for (int k = 0; k < 40; ++k) {
            mu += hl[k] * f1w[o * 40 + k];
            lg += hl[k] * f2w[o * 40 + k];
        }
        float sg = __expf(lg);
        out[b * 32 + o] = mu;                               // mu
        out[2048 + b * 32 + o] = mu + sg * eps[b * 32 + o]; // z
    }
}

// ---------------------------------------------------------------------------
extern "C" void kernel_launch(void* const* d_in, const int* in_sizes, int n_in,
                              void* d_out, int out_size, void* d_ws, size_t ws_size,
                              hipStream_t stream) {
    (void)in_sizes; (void)n_in; (void)out_size; (void)ws_size;
    const float* x       = (const float*)d_in[0];
    const float* eps     = (const float*)d_in[1];
    const float* conv1_w = (const float*)d_in[2];
    const float* conv1_b = (const float*)d_in[3];
    const float* bn1_g   = (const float*)d_in[4];
    const float* bn1_b   = (const float*)d_in[5];
    const float* conv2_w = (const float*)d_in[6];
    const float* conv3_w = (const float*)d_in[8];
    const float* conv3_b = (const float*)d_in[9];
    const float* bn2_g   = (const float*)d_in[10];
    const float* bn2_b   = (const float*)d_in[11];
    const float* d1_wih  = (const float*)d_in[12];
    const float* d1_whh  = (const float*)d_in[13];
    const float* d1_bih  = (const float*)d_in[14];
    const float* d1_bhh  = (const float*)d_in[15];
    const float* d2_wih  = (const float*)d_in[16];
    const float* d2_whh  = (const float*)d_in[17];
    const float* d2_bih  = (const float*)d_in[18];
    const float* d2_bhh  = (const float*)d_in[19];
    const float* d3_wih  = (const float*)d_in[20];
    const float* d3_whh  = (const float*)d_in[21];
    const float* d3_bih  = (const float*)d_in[22];
    const float* d3_bhh  = (const float*)d_in[23];
    const float* d4_wih  = (const float*)d_in[24];
    const float* d4_whh  = (const float*)d_in[25];
    const float* d4_bih  = (const float*)d_in[26];
    const float* d4_bhh  = (const float*)d_in[27];
    const float* d_fc_w  = (const float*)d_in[28];
    const float* d_fc_b  = (const float*)d_in[29];
    const float* w_wih   = (const float*)d_in[30];
    const float* w_whh   = (const float*)d_in[31];
    const float* w_bih   = (const float*)d_in[32];
    const float* w_bhh   = (const float*)d_in[33];
    const float* w_fc_w  = (const float*)d_in[34];
    const float* w_fc_b  = (const float*)d_in[35];
    const float* l1_w    = (const float*)d_in[36];
    const float* l1_b    = (const float*)d_in[37];
    const float* fc1_w   = (const float*)d_in[38];
    const float* fc1_b   = (const float*)d_in[39];
    const float* fc2_w   = (const float*)d_in[40];
    const float* fc2_b   = (const float*)d_in[41];

    char* wsb = (char*)d_ws;
    size_t o = 0;
    auto take = [&](size_t bytes) { size_t r = o; o = (o + bytes + 255) & ~(size_t)255; return r; };
    half_t* P    = (half_t*)(wsb + take((size_t)BSZ * C1N * TP * F1 * 2));
    half_t* Y    = (half_t*)(wsb + take((size_t)BSZ * C1N * TP * F1P * 2));
    float*  st   = (float*)(wsb + take(256 * 4));
    half_t* Wp2  = (half_t*)(wsb + take((size_t)K2P * 48 * 2));
    float*  fpre = (float*)(wsb + take((size_t)BSZ * TT * F2 * 4));
    half_t* feat = (half_t*)(wsb + take((size_t)BSZ * TT * F2 * 2));
    half_t* Wihw = (half_t*)(wsb + take((size_t)256 * 448 * 2));
    half_t* Wih1 = (half_t*)(wsb + take((size_t)256 * 368 * 2));
    float*  Xgw  = (float*)(wsb + take((size_t)8192 * 448 * 4));
    float*  Xg1  = (float*)(wsb + take((size_t)8192 * 368 * 4));
    float*  hwf  = (float*)(wsb + take((size_t)BSZ * 110 * 4));
    float*  h4f  = (float*)(wsb + take((size_t)BSZ * 24 * 4));

    // 1. init pads + stat accumulators
    {
        int total = BSZ * C1N * 2 * F1 + BSZ * C1N * 2 * F1P + 64;
        k_init<<<(total + 255) / 256, 256, 0, stream>>>(P, Y, st);
    }
    // 2. pack weights into f16 GEMM layouts
    k_pack_w2<<<(K2P * 48 + 255) / 256, 256, 0, stream>>>(conv2_w, Wp2);
    k_pack_T<<<(256 * 448 + 255) / 256, 256, 0, stream>>>(w_wih, Wihw, 440, 256, 448);
    k_pack_T<<<(256 * 368 + 255) / 256, 256, 0, stream>>>(d1_wih, Wih1, 360, 256, 368);
    // 3. conv1 + silu
    k_conv1<<<(BSZ * C1N * TT * F1) / 256, 256, 0, stream>>>(x, conv1_w, conv1_b, P);
    // 4. bn1 stats over pooled values
    k_stats1<<<dim3(BSZ, C1N), 256, 0, stream>>>(P, st);
    // 5. pool + bn1 -> Y
    k_bn1apply<<<(BSZ * C1N * TT * F1P) / 256, 256, 0, stream>>>(P, st, bn1_g, bn1_b, Y);
    // 6. conv2 WMMA GEMM + silu + conv3 reduce -> featpre
    k_conv2<<<(BSZ * TT * F2) / 128, 256, 0, stream>>>(Y, Wp2, conv3_w, conv3_b, fpre);
    // 7. bn2 stats + apply + silu -> feat [T][B][256] f16
    k_bn2stats<<<(BSZ * TT * F2) / 2048, 256, 0, stream>>>(fpre, st);
    k_bn2apply<<<(BSZ * TT * F2) / 256, 256, 0, stream>>>(fpre, st, bn2_g, bn2_b, feat);
    // 8. hoisted input-gate WMMA GEMMs for both LSTMs (A tiles via TDM)
    k_gemm<<<dim3(8192 / 128, 448 / 16), 256, 0, stream>>>(feat, Wihw, Xgw, 8192, 256, 448);
    k_gemm<<<dim3(8192 / 128, 368 / 16), 256, 0, stream>>>(feat, Wih1, Xg1, 8192, 256, 368);
    // 9. recurrences (sequential, one persistent workgroup each)
    k_wide_rec<<<1, 1024, 0, stream>>>(Xgw, w_whh, w_bih, w_bhh, hwf);
    k_deep_rec<<<1, 1024, 0, stream>>>(Xg1,
                                       d1_whh, d1_bih, d1_bhh,
                                       d2_wih, d2_whh, d2_bih, d2_bhh,
                                       d3_wih, d3_whh, d3_bih, d3_bhh,
                                       d4_wih, d4_whh, d4_bih, d4_bhh,
                                       h4f);
    // 10. head -> d_out = [mu | z]
    k_head<<<1, 64, 0, stream>>>(h4f, hwf, d_fc_w, d_fc_b, w_fc_w, w_fc_b,
                                 l1_w, l1_b, fc1_w, fc1_b, fc2_w, fc2_b,
                                 eps, (float*)d_out);
}
